// DSDHLoss_28355374088706
// MI455X (gfx1250) — 8-bit matrix operand order repaired and checked by reading, compiled, not run on hardware
//
#include <hip/hip_runtime.h>
#include <hip/hip_bf16.h>

#define BITN   64
#define TN     100000
#define TPAD   100096
#define NCHUNK 1564
#define NCLASS 100
#define NBATCH 128
#define DCCI   3
#define KSLICES 46
#define CHUNKS_PER_SLICE 34
#define NTILES_SYM 10
#define NB_LIK 400
#define NTJ    6250
#define BY_BLOCKS 250
#define BY_PER (TN / BY_BLOCKS)

#define MU_F   1.0f
#define NU_F   1.0f
#define ETA_MU 55.0f

typedef int   v8i __attribute__((ext_vector_type(8)));
typedef float v8f __attribute__((ext_vector_type(8)));
typedef float v2f __attribute__((ext_vector_type(2)));

__constant__ int SYM_TI[NTILES_SYM] = {0,0,0,0,1,1,1,2,2,3};
__constant__ int SYM_TJ[NTILES_SYM] = {0,1,2,3,1,2,3,2,3,3};

__global__ void dsdh_prep(const float* __restrict__ U, const float* __restrict__ Bf,
                          float* __restrict__ Uw, signed char* __restrict__ Bi8) {
  int idx = blockIdx.x * blockDim.x + threadIdx.x;
  if (idx < BITN * TN) {
    Uw[idx] = U[idx];
    int row = idx / TN, col = idx % TN;
    float b = Bf[idx];
    Bi8[row * TPAD + col] = (signed char)((b > 0.f) ? 1 : ((b < 0.f) ? -1 : 0));
  }
  if (idx < BITN * (TPAD - TN)) {
    int row = idx / (TPAD - TN), c = idx % (TPAD - TN);
    Bi8[row * TPAD + TN + c] = 0;
  }
}

__global__ void dsdh_ct(const float* __restrict__ Y, int* __restrict__ ct) {
  int t = blockIdx.x * blockDim.x + threadIdx.x;
  if (t >= TN) return;
  float best = -1.f; int bc = 0;
  for (int c = 0; c < NCLASS; ++c) { float v = Y[c * TN + t]; if (v > best) { best = v; bc = c; } }
  ct[t] = bc;
}

__global__ void dsdh_cb(const float* __restrict__ y, int* __restrict__ cb) {
  int k = threadIdx.x;
  if (k >= NBATCH) return;
  float best = -1.f; int bc = 0;
  for (int c = 0; c < NCLASS; ++c) { float v = y[k * NCLASS + c]; if (v > best) { best = v; bc = c; } }
  cb[k] = bc;
}

__global__ void dsdh_scatter(const float* __restrict__ u, const int* __restrict__ ind,
                             const int* __restrict__ cb, float* __restrict__ Uw,
                             int* __restrict__ ct) {
  int k = threadIdx.x;
  if (k >= NBATCH) return;
  int tix = ind[k];
  for (int k2 = k + 1; k2 < NBATCH; ++k2) if (ind[k2] == tix) return;
  ct[tix] = cb[k];
  for (int i = 0; i < BITN; ++i) Uw[i * TN + tix] = u[k * BITN + i];
}

__global__ void dsdh_zero_small(int* __restrict__ Mi, int* __restrict__ BYi) {
  int i = blockIdx.x * blockDim.x + threadIdx.x;
  if (i < BITN * BITN) Mi[i] = 0;
  if (i < BITN * NCLASS) BYi[i] = 0;
}

__global__ void __launch_bounds__(32)
dsdh_gram(const signed char* __restrict__ Bi8, int* __restrict__ Mi) {
  int tile = blockIdx.x;
  int slice = blockIdx.y;
  int ti = SYM_TI[tile], tj = SYM_TJ[tile];
  int lane = threadIdx.x, half = lane >> 4, l16 = lane & 15;
  const signed char* arow = Bi8 + (ti * 16 + l16) * TPAD + 8 * half;
  const signed char* brow = Bi8 + (tj * 16 + l16) * TPAD + 8 * half;

  auto load2 = [&](int ch, v8i& a, v8i& b) {
    int t0 = ch * 64;
    const int* pa = (const int*)(arow + t0);
    const int* pb = (const int*)(brow + t0);
    a[0]=pa[0];  a[1]=pa[1];  a[2]=pa[4];  a[3]=pa[5];
    a[4]=pa[8];  a[5]=pa[9];  a[6]=pa[12]; a[7]=pa[13];
    b[0]=pb[0];  b[1]=pb[1];  b[2]=pb[4];  b[3]=pb[5];
    b[4]=pb[8];  b[5]=pb[9];  b[6]=pb[12]; b[7]=pb[13];
  };

  v8i acc0 = {0,0,0,0,0,0,0,0};
  v8i acc1 = {0,0,0,0,0,0,0,0};
  int c0 = slice * CHUNKS_PER_SLICE;
  v8i a0, b0, a1, b1, a2, b2, a3, b3;
  load2(c0,     a0, b0);
  load2(c0 + 1, a1, b1);
  int ch = c0 + 2;
#pragma unroll 1
  for (int p = 0; p < 8; ++p) {
    load2(ch,     a2, b2);
    load2(ch + 1, a3, b3);
    acc0 = __builtin_amdgcn_wmma_i32_16x16x64_iu8(true, a0, true, b0, acc0, false, false);
    acc1 = __builtin_amdgcn_wmma_i32_16x16x64_iu8(true, a1, true, b1, acc1, false, false);
    load2(ch + 2, a0, b0);
    load2(ch + 3, a1, b1);
    acc0 = __builtin_amdgcn_wmma_i32_16x16x64_iu8(true, a2, true, b2, acc0, false, false);
    acc1 = __builtin_amdgcn_wmma_i32_16x16x64_iu8(true, a3, true, b3, acc1, false, false);
    ch += 4;
  }
  acc0 = __builtin_amdgcn_wmma_i32_16x16x64_iu8(true, a0, true, b0, acc0, false, false);
  acc1 = __builtin_amdgcn_wmma_i32_16x16x64_iu8(true, a1, true, b1, acc1, false, false);
#pragma unroll
  for (int v = 0; v < 8; ++v) {
    int mrow = ti * 16 + v + 8 * half;
    int ncol = tj * 16 + l16;
    atomicAdd(&Mi[mrow * 64 + ncol], acc0[v] + acc1[v]);
  }
}

__global__ void __launch_bounds__(256)
dsdh_by(const signed char* __restrict__ Bi8, const int* __restrict__ ct,
        int* __restrict__ BYi) {
  __shared__ int lacc[BITN * NCLASS];
  for (int e = threadIdx.x; e < BITN * NCLASS; e += 256) lacc[e] = 0;
  __syncthreads();
  int t0 = blockIdx.x * BY_PER;
  for (int t = t0 + threadIdx.x; t < t0 + BY_PER; t += 256) {
    int c = ct[t];
#pragma unroll
    for (int i = 0; i < BITN; ++i)
      atomicAdd(&lacc[i * NCLASS + c], (int)Bi8[i * TPAD + t]);
  }
  __syncthreads();
  for (int e = threadIdx.x; e < BITN * NCLASS; e += 256) {
    int v = lacc[e];
    if (v) atomicAdd(&BYi[e], v);
  }
}

__global__ void __launch_bounds__(256)
dsdh_solve(const int* __restrict__ Mi, const int* __restrict__ BYi,
           float* __restrict__ Wf, float* __restrict__ Gf) {
  __shared__ float Aug[BITN][130];
  __shared__ float fac[BITN];
  __shared__ float Wl[BITN][NCLASS];
  int tid = threadIdx.x;
  for (int e = tid; e < BITN * BITN; e += 256) {
    int i = e >> 6, j = e & 63;
    int mv = (i <= j) ? Mi[i * 64 + j] : Mi[j * 64 + i];
    Aug[i][j]      = (float)mv + ((i == j) ? (NU_F / MU_F) : 0.f);
    Aug[i][64 + j] = (i == j) ? 1.f : 0.f;
  }
  __syncthreads();
  for (int k = 0; k < BITN; ++k) {
    float invp = 1.f / Aug[k][k];
    float f = (tid < BITN) ? Aug[tid][k] : 0.f;
    __syncthreads();
    if (tid < 128) Aug[k][tid] *= invp;
    if (tid < BITN) fac[tid] = f;
    __syncthreads();
    for (int e = tid; e < BITN * 128; e += 256) {
      int r = e >> 7, col = e & 127;
      if (r != k) Aug[r][col] -= fac[r] * Aug[k][col];
    }
    __syncthreads();
  }
  for (int e = tid; e < BITN * NCLASS; e += 256) {
    int i = e / NCLASS, c = e % NCLASS;
    float s = 0.f;
    for (int j = 0; j < BITN; ++j) s = __builtin_fmaf(Aug[i][64 + j], (float)BYi[j * NCLASS + c], s);
    Wl[i][c] = s; Wf[e] = s;
  }
  __syncthreads();
  for (int e = tid; e < BITN * BITN; e += 256) {
    int i = e >> 6, j = e & 63;
    float s = 0.f;
    for (int c = 0; c < NCLASS; ++c) s = __builtin_fmaf(Wl[i][c], Wl[j][c], s);
    Gf[e] = s;
  }
}

__global__ void __launch_bounds__(256)
dsdh_colupdate(signed char* __restrict__ Bi8, const float* __restrict__ Uw,
               const int* __restrict__ ct, const float* __restrict__ Wf,
               const float* __restrict__ Gf) {
  __shared__ float Gl[BITN][BITN];
  __shared__ float Wl[BITN][NCLASS];
  for (int e = threadIdx.x; e < BITN * BITN; e += 256) Gl[e >> 6][e & 63] = Gf[e];
  for (int e = threadIdx.x; e < BITN * NCLASS; e += 256) Wl[e / NCLASS][e % NCLASS] = Wf[e];
  __syncthreads();
  int t = blockIdx.x * 256 + threadIdx.x;
  if (t >= TN) return;
  int c = ct[t];
  float s[BITN];
#pragma unroll
  for (int i = 0; i < BITN; ++i) s[i] = (float)Bi8[i * TPAD + t];
#pragma unroll
  for (int i = 0; i < BITN; ++i) {
    float p = Wl[i][c] + ETA_MU * Uw[i * TN + t];
    float dot = 0.f;
#pragma unroll
    for (int j = 0; j < BITN; ++j) dot = __builtin_fmaf(Gl[i][j], s[j], dot);
    float val = p - dot + Gl[i][i] * s[i];
    s[i] = (val >= 0.f) ? 1.f : -1.f;
  }
#pragma unroll
  for (int i = 0; i < BITN; ++i) Bi8[i * TPAD + t] = (signed char)((s[i] > 0.f) ? 1 : -1);
}

__global__ void __launch_bounds__(256)
dsdh_lik(const float* __restrict__ u, const float* __restrict__ Uw,
         const int* __restrict__ cb, const int* __restrict__ ct,
         float* __restrict__ partials) {
  __shared__ int cbs[NBATCH];
  __shared__ float red[256];
  if (threadIdx.x < NBATCH) cbs[threadIdx.x] = cb[threadIdx.x];
  __syncthreads();
  int wave = threadIdx.x >> 5, lane = threadIdx.x & 31;
  int half = lane >> 4, n = lane & 15;
  int bi = wave;
  v2f afr[16];
  const float* ur = u + (bi * 16 + n) * BITN;
#pragma unroll
  for (int kk = 0; kk < 16; ++kk) {
    int k0 = kk * 4;
    afr[kk][0] = ur[k0 + 2 * half];
    afr[kk][1] = ur[k0 + 2 * half + 1];
  }
  int myc[8];
#pragma unroll
  for (int v = 0; v < 8; ++v) myc[v] = cbs[bi * 16 + v + 8 * half];
  float lsum = 0.f;
#pragma unroll 1
  for (int tj = blockIdx.x; tj < NTJ; tj += NB_LIK) {
    int t0 = tj * 16;
    int tc = ct[t0 + n];
    v8f acc0 = {0.f,0.f,0.f,0.f,0.f,0.f,0.f,0.f};
    v8f acc1 = {0.f,0.f,0.f,0.f,0.f,0.f,0.f,0.f};
#pragma unroll
    for (int kk = 0; kk < 16; kk += 2) {
      int k0 = kk * 4;
      v2f bfr0, bfr1;
      bfr0[0] = Uw[(k0 + 2 * half) * TN + t0 + n];
      bfr0[1] = Uw[(k0 + 2 * half + 1) * TN + t0 + n];
      bfr1[0] = Uw[(k0 + 4 + 2 * half) * TN + t0 + n];
      bfr1[1] = Uw[(k0 + 4 + 2 * half + 1) * TN + t0 + n];
      acc0 = __builtin_amdgcn_wmma_f32_16x16x4_f32(false, afr[kk],     false, bfr0,
                                                   (short)0, acc0, false, false);
      acc1 = __builtin_amdgcn_wmma_f32_16x16x4_f32(false, afr[kk + 1], false, bfr1,
                                                   (short)0, acc1, false, false);
    }
#pragma unroll
    for (int v = 0; v < 8; ++v) {
      float ip = 0.5f * (acc0[v] + acc1[v]);
      float lik = log1pf(expf(-fabsf(ip))) + fmaxf(ip, 0.f);
      if (myc[v] == tc) lik -= ip;
      lsum += lik;
    }
  }
  red[threadIdx.x] = lsum;
  __syncthreads();
  for (int s2 = 128; s2 > 0; s2 >>= 1) {
    if (threadIdx.x < s2) red[threadIdx.x] += red[threadIdx.x + s2];
    __syncthreads();
  }
  if (threadIdx.x == 0) partials[blockIdx.x] = red[0];
}

__global__ void __launch_bounds__(256)
dsdh_final(const float* __restrict__ y, const int* __restrict__ ind,
           const signed char* __restrict__ Bi8, const float* __restrict__ Wf,
           const float* __restrict__ partials, float* __restrict__ out) {
  __shared__ float Wl[BITN * NCLASS];
  __shared__ float red[256];
  int tid = threadIdx.x;
  for (int e = tid; e < BITN * NCLASS; e += 256) Wl[e] = Wf[e];
  __syncthreads();
  float liks = 0.f;
  for (int e = tid; e < NB_LIK; e += 256) liks += partials[e];
  float cls = 0.f;
  for (int e = tid; e < NCLASS * NBATCH; e += 256) {
    int c = e / NBATCH, b = e % NBATCH;
    int tcol = ind[b];
    float pred = 0.f;
#pragma unroll
    for (int i = 0; i < BITN; ++i)
      pred = __builtin_fmaf(Wl[i * NCLASS + c], (float)Bi8[i * TPAD + tcol], pred);
    float d = y[b * NCLASS + c] - pred;
    cls += d * d;
  }
  float regs = 0.f;
  for (int e = tid; e < BITN * NCLASS; e += 256) { float w = Wl[e]; regs += w * w; }

  red[tid] = liks; __syncthreads();
  for (int s2 = 128; s2 > 0; s2 >>= 1) { if (tid < s2) red[tid] += red[tid + s2]; __syncthreads(); }
  float tl = red[0]; __syncthreads();
  red[tid] = cls; __syncthreads();
  for (int s2 = 128; s2 > 0; s2 >>= 1) { if (tid < s2) red[tid] += red[tid + s2]; __syncthreads(); }
  float tc = red[0]; __syncthreads();
  red[tid] = regs; __syncthreads();
  for (int s2 = 128; s2 > 0; s2 >>= 1) { if (tid < s2) red[tid] += red[tid + s2]; __syncthreads(); }
  if (tid == 0) {
    float tr = red[0];
    out[0] = tl / (float)(NBATCH * (long long)TN)
           + MU_F * (tc / (float)(NCLASS * NBATCH))
           + NU_F * (tr / (float)(BITN * NCLASS));
  }
}

extern "C" void kernel_launch(void* const* d_in, const int* in_sizes, int n_in,
                              void* d_out, int out_size, void* d_ws, size_t ws_size,
                              hipStream_t stream) {
  const float* u   = (const float*)d_in[0];
  const float* y   = (const float*)d_in[1];
  const float* U   = (const float*)d_in[2];
  const float* Bf  = (const float*)d_in[3];
  const float* Y   = (const float*)d_in[4];
  const int*   ind = (const int*)d_in[5];

  char* ws = (char*)d_ws;
  float*       Uw   = (float*)ws;
  signed char* Bi8  = (signed char*)(ws + 25600000);
  int*         ct   = (int*)(ws + 32006144);
  int*         cb   = ct + TN;
  int*         Mi   = cb + NBATCH;
  int*         BYi  = Mi + BITN * BITN;
  float*       Wf   = (float*)(BYi + BITN * NCLASS);
  float*       Gf   = Wf + BITN * NCLASS;
  float*       part = Gf + BITN * BITN;

  dsdh_prep<<<(BITN * TN + 255) / 256, 256, 0, stream>>>(U, Bf, Uw, Bi8);
  dsdh_ct<<<(TN + 255) / 256, 256, 0, stream>>>(Y, ct);
  dsdh_cb<<<1, 128, 0, stream>>>(y, cb);
  dsdh_scatter<<<1, 128, 0, stream>>>(u, ind, cb, Uw, ct);

  for (int it = 0; it < DCCI; ++it) {
    dsdh_zero_small<<<26, 256, 0, stream>>>(Mi, BYi);
    dsdh_gram<<<dim3(NTILES_SYM, KSLICES), 32, 0, stream>>>(Bi8, Mi);
    dsdh_by<<<BY_BLOCKS, 256, 0, stream>>>(Bi8, ct, BYi);
    dsdh_solve<<<1, 256, 0, stream>>>(Mi, BYi, Wf, Gf);
    dsdh_colupdate<<<(TN + 255) / 256, 256, 0, stream>>>(Bi8, Uw, ct, Wf, Gf);
  }
  dsdh_lik<<<NB_LIK, 256, 0, stream>>>(u, Uw, cb, ct, part);
  dsdh_final<<<1, 256, 0, stream>>>(y, ind, Bi8, Wf, part, (float*)d_out);
}
